// MultiHeadSelfAttention_48447231099295
// MI455X (gfx1250) — matrix operand order repaired and checked by reading, not compiled, not run on hardware
//
#include <hip/hip_runtime.h>
#include <hip/hip_bf16.h>

typedef __bf16 bf16;
typedef __attribute__((ext_vector_type(16))) __bf16 v16bf;
typedef __attribute__((ext_vector_type(8)))  __bf16 bf16x8;
typedef __attribute__((ext_vector_type(2)))  __bf16 bf16x2;
typedef __attribute__((ext_vector_type(8)))  float  v8f;

#define D_MODEL   1024
#define NUM_HEADS 16
#define D_K       64
#define SEQ       2048
#define BATCH     4

static __device__ __forceinline__ v8f wmma_bf16(v16bf a, v16bf b, v8f c) {
  // D = A(16x32 bf16) * B(32x16 bf16) + C(16x16 f32)
  return __builtin_amdgcn_wmma_f32_16x16x32_bf16(false, a, false, b, (short)0, c,
                                                 false, false);
}

union AFrag {
  v16bf  v;
  bf16x2 p[8];  // p[v] = the two halves (K=2v, 2v+1 pattern) of VGPR v
};

// ---------------------------------------------------------------------------
// Stage 0: one-shot f32 -> bf16 conversion (x and the four weight matrices).
// Keeps all cvt VALU work out of the GEMM inner loops.
// ---------------------------------------------------------------------------
__global__ __launch_bounds__(256)
void cvt_f32_bf16_kernel(const float* __restrict__ src, bf16* __restrict__ dst,
                         int n) {
  const int i = (blockIdx.x * 256 + threadIdx.x) * 8;
  if (i >= n) return;
  const float4 a = *(const float4*)(src + i);
  const float4 b = *(const float4*)(src + i + 4);
  bf16x8 o;
  o[0] = (bf16)a.x; o[1] = (bf16)a.y; o[2] = (bf16)a.z; o[3] = (bf16)a.w;
  o[4] = (bf16)b.x; o[5] = (bf16)b.y; o[6] = (bf16)b.z; o[7] = (bf16)b.w;
  *(bf16x8*)(dst + i) = o;
}

// ---------------------------------------------------------------------------
// Stage 1: Q/K/V = x @ W^T (bf16 WMMA), RoPE on Q/K, 1/sqrt(d_k) folded into Q.
// Each wave: 32(M) x 64(N) output tile, K-loop of 32 -> 8 WMMAs per step.
// Q,K stored [B,H,S,64] bf16; V stored transposed [B,H,64,S] bf16.
// grid: (head=16, row_tile=256, mat=3), block 32.
// ---------------------------------------------------------------------------
__global__ __launch_bounds__(32)
void qkv_rope_kernel(const bf16*  __restrict__ xb,
                     const int*   __restrict__ tpos,
                     const bf16*  __restrict__ Wqb,
                     const bf16*  __restrict__ Wkb,
                     const bf16*  __restrict__ Wvb,
                     bf16* __restrict__ Qo,
                     bf16* __restrict__ Ko,
                     bf16* __restrict__ Vto)
{
  const int lane = threadIdx.x;
  const int g    = lane >> 4;     // half-wave group
  const int ln   = lane & 15;
  const int head = blockIdx.x;    // 64-column tile == one head
  const int mt   = blockIdx.y;    // 32-row tile over B*S
  const int mat  = blockIdx.z;    // 0=Q 1=K 2=V
  const bf16* W  = (mat == 0) ? Wqb : ((mat == 1) ? Wkb : Wvb);
  const int row0 = mt * 32;

  v8f acc[2][4] = {};
  const bf16* xr0 = xb + (size_t)(row0 + ln) * D_MODEL;
  const bf16* xr1 = xb + (size_t)(row0 + 16 + ln) * D_MODEL;

  for (int ks = 0; ks < D_MODEL; ks += 32) {
    // A fragments: M = ln, K = ks + g*16 + hi*8 + mid*2 + lo
    AFrag a0, a1;
#pragma unroll
    for (int hi = 0; hi < 2; ++hi)
#pragma unroll
      for (int mid = 0; mid < 4; ++mid) {
        const int d = ks + g * 16 + hi * 8 + mid * 2;
        a0.p[hi * 4 + mid] = *(const bf16x2*)(xr0 + d);
        a1.p[hi * 4 + mid] = *(const bf16x2*)(xr1 + d);
      }
#pragma unroll
    for (int cc = 0; cc < 4; ++cc) {
      // B fragment: K = ks + g*16 + h, N = ln ; B[k][n] = W[n][k]
      const int n = head * 64 + cc * 16 + ln;
      const v16bf bfr = *(const v16bf*)(W + (size_t)n * D_MODEL + ks + g * 16);
      acc[0][cc] = wmma_bf16(a0.v, bfr, acc[0][cc]);
      acc[1][cc] = wmma_bf16(a1.v, bfr, acc[1][cc]);
    }
  }

  const int b_idx = row0 / SEQ;
  const int s0    = row0 % SEQ;

  // token positions for the 16 rows this lane-half contributes to
  float posf[2][8];
#pragma unroll
  for (int rg = 0; rg < 2; ++rg)
#pragma unroll
    for (int r = 0; r < 8; ++r)
      posf[rg][r] = (float)tpos[b_idx * SEQ + s0 + rg * 16 + r + 8 * g];

#pragma unroll
  for (int rg = 0; rg < 2; ++rg) {
#pragma unroll
    for (int cc = 0; cc < 4; ++cc) {
      const int dl = cc * 16 + ln;          // dim within head, 0..63
      if (mat < 2) {
        // RoPE: pairs (2i,2i+1); inv_freq = theta^(-i/32) = exp(-i*ln(1e4)/32)
        const int   i        = dl >> 1;
        const float inv_freq = __expf((float)i * -0.28782313662425572f);
#pragma unroll
        for (int r = 0; r < 8; ++r) {
          const float ang = posf[rg][r] * inv_freq;
          const float cs  = __cosf(ang), sn = __sinf(ang);
          const float own     = acc[rg][cc][r];
          const float partner = __shfl_xor(own, 1, 32);  // even<->odd column
          acc[rg][cc][r] = ((ln & 1) == 0) ? (own * cs - partner * sn)
                                           : (partner * sn + own * cs);
        }
      }
#pragma unroll
      for (int r = 0; r < 8; ++r) {
        const int s = s0 + rg * 16 + r + 8 * g;   // C layout: M = r + 8*g
        float v = acc[rg][cc][r];
        if (mat == 0) {
          v *= 0.125f;  // 1/sqrt(D_K) folded into Q
          Qo[(((size_t)b_idx * NUM_HEADS + head) * SEQ + s) * D_K + dl] = (bf16)v;
        } else if (mat == 1) {
          Ko[(((size_t)b_idx * NUM_HEADS + head) * SEQ + s) * D_K + dl] = (bf16)v;
        } else {
          Vto[(((size_t)b_idx * NUM_HEADS + head) * D_K + dl) * SEQ + s] = (bf16)v;
        }
      }
    }
  }
}

// ---------------------------------------------------------------------------
// Stage 2: causal flash attention, one wave per (b, h, 16-query tile).
// grid: (S/16=128, H=16, B=4), block 32.
// ---------------------------------------------------------------------------
__global__ __launch_bounds__(32)
void attention_kernel(const bf16* __restrict__ Q,
                      const bf16* __restrict__ K,
                      const bf16* __restrict__ Vt,
                      bf16* __restrict__ Comb)
{
  __shared__ bf16 Plds[16 * 32];  // P tile staging (C-frag -> A-frag relayout)

  const int lane = threadIdx.x;
  const int g    = lane >> 4;
  const int ln   = lane & 15;
  const int qt   = blockIdx.x;
  const int h    = blockIdx.y;
  const int b    = blockIdx.z;
  const size_t bh = (size_t)b * NUM_HEADS + h;
  const bf16* Qp = Q  + bh * SEQ * D_K;
  const bf16* Kp = K  + bh * SEQ * D_K;
  const bf16* Vp = Vt + bh * D_K * SEQ;

  // Query A fragments (d = 0..31 and 32..63); M = ln
  AFrag a0, a1;
  const int sA = qt * 16 + ln;
#pragma unroll
  for (int hi = 0; hi < 2; ++hi)
#pragma unroll
    for (int mid = 0; mid < 4; ++mid) {
      const int d = g * 16 + hi * 8 + mid * 2;
      a0.p[hi * 4 + mid] = *(const bf16x2*)(Qp + (size_t)sA * D_K + d);
      a1.p[hi * 4 + mid] = *(const bf16x2*)(Qp + (size_t)sA * D_K + 32 + d);
    }

  float m_i[8], l_i[8];
  v8f   o[4] = {};
#pragma unroll
  for (int r = 0; r < 8; ++r) { m_i[r] = -1e30f; l_i[r] = 0.0f; }

  const int nchunks = (qt * 16 + 16 + 31) / 32;  // keys up to causal diagonal
  for (int kc = 0; kc < nchunks; ++kc) {
    const int keybase = kc * 32;

    // prefetch next chunk's K rows / V columns (global_prefetch_b8)
    if (kc + 1 < nchunks) {
      __builtin_prefetch(Kp + (size_t)(keybase + 32 + lane) * D_K, 0, 1);
      __builtin_prefetch(Vp + (size_t)lane * SEQ + keybase + 32, 0, 1);
      __builtin_prefetch(Vp + (size_t)(lane + 32) * SEQ + keybase + 32, 0, 1);
    }

    // ---- scores: S = Q * K^T (pre-scaled), two 16-key column tiles ----
    v8f sc[2];
#pragma unroll
    for (int nt2 = 0; nt2 < 2; ++nt2) {
      const int key = keybase + nt2 * 16 + ln;
      const bf16* krow = Kp + (size_t)key * D_K;
      const v16bf blo = *(const v16bf*)(krow + g * 16);       // d = g*16 + h
      const v16bf bhi = *(const v16bf*)(krow + 32 + g * 16);
      v8f c = {};
      c = wmma_bf16(a0.v, blo, c);
      c = wmma_bf16(a1.v, bhi, c);
#pragma unroll
      for (int r = 0; r < 8; ++r) {          // causal mask
        const int qrow = qt * 16 + r + 8 * g;
        if (key > qrow) c[r] = -1e30f;
      }
      sc[nt2] = c;
    }

    // ---- online softmax (rows live across the 16 lanes of each half) ----
    float alpha[8], rsum[8];
#pragma unroll
    for (int r = 0; r < 8; ++r) {
      float rm = fmaxf(sc[0][r], sc[1][r]);
#pragma unroll
      for (int off = 1; off < 16; off <<= 1)
        rm = fmaxf(rm, __shfl_xor(rm, off, 32));
      const float mnew = fmaxf(m_i[r], rm);
      alpha[r] = __expf(m_i[r] - mnew);
      m_i[r]   = mnew;
      rsum[r]  = 0.0f;
    }
#pragma unroll
    for (int nt2 = 0; nt2 < 2; ++nt2)
#pragma unroll
      for (int r = 0; r < 8; ++r) {
        const float p = __expf(sc[nt2][r] - m_i[r]);
        rsum[r] += p;
        Plds[(r + 8 * g) * 32 + nt2 * 16 + ln] = (bf16)p;  // row-major 16x32
      }
#pragma unroll
    for (int r = 0; r < 8; ++r) {
      float rs = rsum[r];
#pragma unroll
      for (int off = 1; off < 16; off <<= 1)
        rs += __shfl_xor(rs, off, 32);
      l_i[r] = l_i[r] * alpha[r] + rs;
    }
#pragma unroll
    for (int dt = 0; dt < 4; ++dt)
#pragma unroll
      for (int r = 0; r < 8; ++r)
        o[dt][r] *= alpha[r];

    // ---- P back as A fragment (same-wave LDS ops are in order) ----
    AFrag ap;
#pragma unroll
    for (int hi = 0; hi < 2; ++hi)
#pragma unroll
      for (int mid = 0; mid < 4; ++mid) {
        const int kk = g * 16 + hi * 8 + mid * 2;
        ap.p[hi * 4 + mid] = *(const bf16x2*)(&Plds[ln * 32 + kk]);
      }

    // ---- O += P * V  (V transposed: contiguous over keys) ----
#pragma unroll
    for (int dt = 0; dt < 4; ++dt) {
      const int d = dt * 16 + ln;
      const v16bf bv = *(const v16bf*)(Vp + (size_t)d * SEQ + keybase + g * 16);
      o[dt] = wmma_bf16(ap.v, bv, o[dt]);
    }
  }

  // epilogue: divide by softmax denominator, write combined [B,S,H*64]
#pragma unroll
  for (int dt = 0; dt < 4; ++dt)
#pragma unroll
    for (int r = 0; r < 8; ++r) {
      const int s = qt * 16 + r + 8 * g;
      const float v = o[dt][r] / l_i[r];
      Comb[((size_t)b * SEQ + s) * D_MODEL + h * D_K + dt * 16 + ln] = (bf16)v;
    }
}

// ---------------------------------------------------------------------------
// Stage 3: out = combined @ Wo^T, f32 output. 32(M) x 64(N) per wave.
// grid: (16, 256), block 32.
// ---------------------------------------------------------------------------
__global__ __launch_bounds__(32)
void out_proj_kernel(const bf16* __restrict__ Comb,
                     const bf16* __restrict__ Wob,
                     float* __restrict__ Out)
{
  const int lane = threadIdx.x;
  const int g    = lane >> 4;
  const int ln   = lane & 15;
  const int nt   = blockIdx.x;
  const int mt   = blockIdx.y;
  const int row0 = mt * 32;

  v8f acc[2][4] = {};
  const bf16* cr0 = Comb + (size_t)(row0 + ln) * D_MODEL;
  const bf16* cr1 = Comb + (size_t)(row0 + 16 + ln) * D_MODEL;

  for (int ks = 0; ks < D_MODEL; ks += 32) {
    AFrag a0, a1;
#pragma unroll
    for (int hi = 0; hi < 2; ++hi)
#pragma unroll
      for (int mid = 0; mid < 4; ++mid) {
        const int d = ks + g * 16 + hi * 8 + mid * 2;
        a0.p[hi * 4 + mid] = *(const bf16x2*)(cr0 + d);
        a1.p[hi * 4 + mid] = *(const bf16x2*)(cr1 + d);
      }
#pragma unroll
    for (int cc = 0; cc < 4; ++cc) {
      const int n = nt * 64 + cc * 16 + ln;
      const v16bf bfr = *(const v16bf*)(Wob + (size_t)n * D_MODEL + ks + g * 16);
      acc[0][cc] = wmma_bf16(a0.v, bfr, acc[0][cc]);
      acc[1][cc] = wmma_bf16(a1.v, bfr, acc[1][cc]);
    }
  }
#pragma unroll
  for (int rg = 0; rg < 2; ++rg)
#pragma unroll
    for (int cc = 0; cc < 4; ++cc)
#pragma unroll
      for (int r = 0; r < 8; ++r)
        Out[(size_t)(row0 + rg * 16 + r + 8 * g) * D_MODEL + nt * 64 +
            cc * 16 + ln] = acc[rg][cc][r];
}

// ---------------------------------------------------------------------------
extern "C" void kernel_launch(void* const* d_in, const int* in_sizes, int n_in,
                              void* d_out, int out_size, void* d_ws,
                              size_t ws_size, hipStream_t stream) {
  const float* x  = (const float*)d_in[0];
  const int*   tp = (const int*)d_in[1];
  const float* Wq = (const float*)d_in[2];
  const float* Wk = (const float*)d_in[3];
  const float* Wv = (const float*)d_in[4];
  const float* Wo = (const float*)d_in[5];
  float* out = (float*)d_out;

  const size_t tok_elems = (size_t)BATCH * SEQ * D_MODEL;   // 8 Mi
  const size_t w_elems   = (size_t)D_MODEL * D_MODEL;       // 1 Mi

  bf16* Qw   = (bf16*)d_ws;           // 16 MB
  bf16* Kw   = Qw  + tok_elems;       // 16 MB
  bf16* Vtw  = Kw  + tok_elems;       // 16 MB (transposed V)
  bf16* Xbf  = Vtw + tok_elems;       // 16 MB; dead after stage 1 ...
  bf16* Comb = Xbf;                   // ... reused for attention output
  bf16* Wqb  = Xbf + tok_elems;       // 2 MB
  bf16* Wkb  = Wqb + w_elems;         // 2 MB
  bf16* Wvb  = Wkb + w_elems;         // 2 MB
  bf16* Wob  = Wvb + w_elems;         // 2 MB

  // Stage 0: one-shot bf16 conversion
  cvt_f32_bf16_kernel<<<(int)(tok_elems / 8 / 256), 256, 0, stream>>>(
      x, Xbf, (int)tok_elems);
  cvt_f32_bf16_kernel<<<(int)(w_elems / 8 / 256), 256, 0, stream>>>(
      Wq, Wqb, (int)w_elems);
  cvt_f32_bf16_kernel<<<(int)(w_elems / 8 / 256), 256, 0, stream>>>(
      Wk, Wkb, (int)w_elems);
  cvt_f32_bf16_kernel<<<(int)(w_elems / 8 / 256), 256, 0, stream>>>(
      Wv, Wvb, (int)w_elems);
  cvt_f32_bf16_kernel<<<(int)(w_elems / 8 / 256), 256, 0, stream>>>(
      Wo, Wob, (int)w_elems);

  qkv_rope_kernel<<<dim3(NUM_HEADS, (BATCH * SEQ) / 32, 3), 32, 0, stream>>>(
      Xbf, tp, Wqb, Wkb, Wvb, Qw, Kw, Vtw);
  attention_kernel<<<dim3(SEQ / 16, NUM_HEADS, BATCH), 32, 0, stream>>>(
      Qw, Kw, Vtw, Comb);
  out_proj_kernel<<<dim3(NUM_HEADS, (BATCH * SEQ) / 32), 32, 0, stream>>>(
      Comb, Wob, out);
}
